// ImageLevelContext_66468913873237
// MI455X (gfx1250) — matrix operand order repaired and checked by reading, not compiled, not run on hardware
//
#include <hip/hip_runtime.h>

typedef __attribute__((ext_vector_type(16))) __bf16 v16bf;
typedef __attribute__((ext_vector_type(8)))  __bf16 v8bf;
typedef __attribute__((ext_vector_type(8)))  float  v8f;

#define N_TOT 8192   // D*H*W = 8*32*32

#if defined(__has_builtin)
#if __has_builtin(__builtin_amdgcn_global_load_async_to_lds_b128)
#define HAVE_ASYNC_LDS 1
#endif
#if __has_builtin(__builtin_amdgcn_permlane16)
#define HAVE_PERMLANE16 1
#endif
#endif

// 16-byte global -> LDS copy: async on CDNA5 (ASYNCcnt), sync fallback otherwise.
__device__ __forceinline__ void cp16_g2l(void* lds, const void* g) {
#ifdef HAVE_ASYNC_LDS
  typedef int gv4i __attribute__((vector_size(16)));
  __builtin_amdgcn_global_load_async_to_lds_b128((gv4i*)g, (gv4i*)lds, 0, 0);
#else
  *(uint4*)lds = *(const uint4*)g;
#endif
}
__device__ __forceinline__ void async_wait_all() {
#ifdef HAVE_ASYNC_LDS
#if __has_builtin(__builtin_amdgcn_s_wait_asynccnt)
  __builtin_amdgcn_s_wait_asynccnt(0);
#else
  asm volatile("s_wait_asynccnt 0" ::: "memory");
#endif
#endif
}

__device__ __forceinline__ float leaky(float v) { return v > 0.f ? v : 0.1f * v; }

#ifdef HAVE_PERMLANE16
__device__ __forceinline__ float permf16(float v, unsigned s0, unsigned s1) {
  int x = __float_as_int(v);
  int y = __builtin_amdgcn_permlane16(x, x, (int)s0, (int)s1, false, false);
  return __int_as_float(y);
}
#endif
// max across each 16-lane half-wave group (VALU permlane path, LDS-free)
__device__ __forceinline__ float rowmax16(float v) {
#ifdef HAVE_PERMLANE16
  v = fmaxf(v, permf16(v, 0x67452301u, 0xEFCDAB89u));  // xor 1
  v = fmaxf(v, permf16(v, 0x54761032u, 0xDCFE98BAu));  // xor 2
  v = fmaxf(v, permf16(v, 0x32107654u, 0xBA98FEDCu));  // xor 4
  v = fmaxf(v, permf16(v, 0xFEDCBA98u, 0x76543210u));  // xor 8
#else
#pragma unroll
  for (int off = 1; off < 16; off <<= 1) v = fmaxf(v, __shfl_xor(v, off, 32));
#endif
  return v;
}

// A-fragment builder: two 16B chunks -> v16bf
__device__ __forceinline__ v16bf mk_afrag(const __bf16* p0, const __bf16* p1) {
  v8bf lo = *(const v8bf*)p0;
  v8bf hi = *(const v8bf*)p1;
  v16bf r;
#pragma unroll
  for (int i = 0; i < 8; ++i) { r[i] = lo[i]; r[i + 8] = hi[i]; }
  return r;
}

// ---------------------------------------------------------------- channel mean
__global__ void k_mean(const float* __restrict__ x, float* __restrict__ mean) {
  __shared__ float red[256];
  const int c = blockIdx.x;
  const float* p = x + (size_t)c * N_TOT;
  float s = 0.f;
  for (int i = threadIdx.x; i < N_TOT; i += 256) s += p[i];
  red[threadIdx.x] = s;
  __syncthreads();
  for (int off = 128; off > 0; off >>= 1) {
    if (threadIdx.x < off) red[threadIdx.x] += red[threadIdx.x + off];
    __syncthreads();
  }
  if (threadIdx.x == 0) mean[c] = red[0] * (1.f / N_TOT);
}

// ------------------------------------------------- fused q/k/v projections
__global__ void k_proj(const float* __restrict__ x, const float* __restrict__ mean,
                       const float* __restrict__ wq1, const float* __restrict__ sq1, const float* __restrict__ bq1,
                       const float* __restrict__ wq2, const float* __restrict__ sq2, const float* __restrict__ bq2,
                       const float* __restrict__ wk1, const float* __restrict__ sk1, const float* __restrict__ bk1,
                       const float* __restrict__ wk2, const float* __restrict__ sk2, const float* __restrict__ bk2,
                       const float* __restrict__ wv,  const float* __restrict__ sv,  const float* __restrict__ bv,
                       __bf16* __restrict__ qbf, __bf16* __restrict__ kbf, __bf16* __restrict__ vbf) {
  __shared__ float xs[64 * 128];
  __shared__ float q1[64 * 128];
  __shared__ float k1[64 * 128];
  __shared__ float cmK[64], cmV[64];
  const int tid = threadIdx.x;
  const int n0 = blockIdx.x * 128;

  for (int i = tid; i < 64 * 32; i += 256) {
    int c = i >> 5, ch4 = i & 31;
    ((float4*)xs)[c * 32 + ch4] = ((const float4*)(x + (size_t)c * N_TOT + n0))[ch4];
  }
  if (tid < 64) {
    float s = 0.f;
    for (int c = 0; c < 64; ++c) s += wk1[tid * 128 + c] * mean[c];
    cmK[tid] = s;
  } else if (tid < 128) {
    int t = tid - 64;
    float s = 0.f;
    for (int c = 0; c < 64; ++c) s += wv[t * 128 + c] * mean[c];
    cmV[t] = s;
  }
  __syncthreads();

  for (int i = tid; i < 3 * 64 * 128; i += 256) {
    int kind = i / (64 * 128);
    int r = i - kind * (64 * 128);
    int t = r >> 7, p = r & 127;
    if (kind == 0) {
      float acc = 0.f;
      for (int c = 0; c < 64; ++c) acc += wq1[t * 64 + c] * xs[c * 128 + p];
      q1[t * 128 + p] = leaky(sq1[t] * acc + bq1[t]);
    } else if (kind == 1) {
      float acc = cmK[t];
      for (int c = 0; c < 64; ++c) acc += wk1[t * 128 + 64 + c] * xs[c * 128 + p];
      k1[t * 128 + p] = leaky(sk1[t] * acc + bk1[t]);
    } else {
      float acc = cmV[t];
      for (int c = 0; c < 64; ++c) acc += wv[t * 128 + 64 + c] * xs[c * 128 + p];
      vbf[(size_t)t * N_TOT + n0 + p] = (__bf16)leaky(sv[t] * acc + bv[t]);
    }
  }
  __syncthreads();

  for (int i = tid; i < 2 * 64 * 128; i += 256) {
    int kind = i / (64 * 128);
    int r = i - kind * (64 * 128);
    int t = r >> 7, p = r & 127;
    float acc = 0.f;
    if (kind == 0) {
      for (int c = 0; c < 64; ++c) acc += wq2[t * 64 + c] * q1[c * 128 + p];
      qbf[(size_t)(n0 + p) * 64 + t] = (__bf16)(leaky(sq2[t] * acc + bq2[t]) * 0.125f);
    } else {
      for (int c = 0; c < 64; ++c) acc += wk2[t * 64 + c] * k1[c * 128 + p];
      kbf[(size_t)(n0 + p) * 64 + t] = (__bf16)leaky(sk2[t] * acc + bk2[t]);
    }
  }
}

// -------------------------------------------------------- flash attention
// double-buffered async K/V tiles; row sums via P @ ones WMMA; permlane row max
__global__ __launch_bounds__(256, 2) void k_attn(
    const __bf16* __restrict__ qbf, const __bf16* __restrict__ kbf,
    const __bf16* __restrict__ vbf, float* __restrict__ ctx) {
  __shared__ __align__(64) __bf16 qs[128 * 64];
  __shared__ __align__(64) __bf16 kt[2][64 * 64];     // key-major
  __shared__ __align__(64) __bf16 vt[2][64 * 64];     // t-major
  __shared__ __align__(64) __bf16 pbuf[8 * 16 * 64];
  const int tid  = threadIdx.x;
  const int wave = tid >> 5;
  const int lane = tid & 31;
  const int ln   = lane & 15;
  const int hi   = lane >> 4;
  const int row = tid >> 2, chunk = tid & 3;

  auto load_tiles = [&](int buf, int kbase) {
    cp16_g2l(&kt[buf][row * 64 + chunk * 16],     kbf + (size_t)(kbase + row) * 64 + chunk * 16);
    cp16_g2l(&kt[buf][row * 64 + chunk * 16 + 8], kbf + (size_t)(kbase + row) * 64 + chunk * 16 + 8);
    cp16_g2l(&vt[buf][row * 64 + chunk * 16],     vbf + (size_t)row * N_TOT + kbase + chunk * 16);
    cp16_g2l(&vt[buf][row * 64 + chunk * 16 + 8], vbf + (size_t)row * N_TOT + kbase + chunk * 16 + 8);
  };

  // stage Q block + first K/V tiles
  {
    const uint4* qsrc = (const uint4*)(qbf + (size_t)blockIdx.x * 128 * 64);
    for (int i = tid; i < 1024; i += 256) cp16_g2l(((uint4*)qs) + i, qsrc + i);
  }
  load_tiles(0, 0);
  async_wait_all();
  __syncthreads();

  v8f oacc[4];
#pragma unroll
  for (int c = 0; c < 4; ++c) oacc[c] = (v8f){};
  float run_max[8], run_sum[8];
#pragma unroll
  for (int j = 0; j < 8; ++j) { run_max[j] = -3.0e38f; run_sum[j] = 0.f; }

  v16bf onesb;
#pragma unroll
  for (int i = 0; i < 16; ++i) onesb[i] = (__bf16)1.0f;

  __bf16* pw = pbuf + wave * (16 * 64);

  for (int it = 0; it < N_TOT / 64; ++it) {
    const int cur = it & 1;
    if (it + 1 < N_TOT / 64) load_tiles(cur ^ 1, (it + 1) * 64);  // async prefetch

    const __bf16* ktc = kt[cur];
    const __bf16* vtc = vt[cur];

    const __bf16* qr = qs + (wave * 16 + ln) * 64;
    v16bf a0 = mk_afrag(qr + hi * 16,      qr + hi * 16 + 8);
    v16bf a1 = mk_afrag(qr + 32 + hi * 16, qr + 32 + hi * 16 + 8);

    // S = Q @ K^T (Q pre-scaled by T^-0.5)
    v8f s[4];
#pragma unroll
    for (int c = 0; c < 4; ++c) {
      const __bf16* kp = ktc + (c * 16 + ln) * 64;
      v16bf b0 = *(const v16bf*)(kp + hi * 16);
      v16bf b1 = *(const v16bf*)(kp + 32 + hi * 16);
      v8f acc = (v8f){};
      acc = __builtin_amdgcn_wmma_f32_16x16x32_bf16(false, a0, false, b0, (short)0, acc, false, false);
      acc = __builtin_amdgcn_wmma_f32_16x16x32_bf16(false, a1, false, b1, (short)0, acc, false, false);
      s[c] = acc;
    }

    // online softmax (max via permlane; sum via WMMA below)
#pragma unroll
    for (int j = 0; j < 8; ++j) {
      float m = fmaxf(fmaxf(s[0][j], s[1][j]), fmaxf(s[2][j], s[3][j]));
      m = rowmax16(m);
      float nm = fmaxf(run_max[j], m);
      float corr = __expf(run_max[j] - nm);
      run_max[j] = nm;
      run_sum[j] *= corr;
#pragma unroll
      for (int c = 0; c < 4; ++c) s[c][j] = __expf(s[c][j] - nm);
#pragma unroll
      for (int c = 0; c < 4; ++c) oacc[c][j] *= corr;
    }

    // relayout P (C-layout -> A-layout) via per-wave LDS slot
#pragma unroll
    for (int c = 0; c < 4; ++c)
#pragma unroll
      for (int j = 0; j < 8; ++j)
        pw[(j + hi * 8) * 64 + c * 16 + ln] = (__bf16)s[c][j];

    const __bf16* prow = pw + ln * 64;
    v16bf pa0 = mk_afrag(prow + hi * 16,      prow + hi * 16 + 8);
    v16bf pa1 = mk_afrag(prow + 32 + hi * 16, prow + 32 + hi * 16 + 8);

    // row sums: lsum = P @ ones  (every column = row sum)
    v8f lsum = (v8f){};
    lsum = __builtin_amdgcn_wmma_f32_16x16x32_bf16(false, pa0, false, onesb, (short)0, lsum, false, false);
    lsum = __builtin_amdgcn_wmma_f32_16x16x32_bf16(false, pa1, false, onesb, (short)0, lsum, false, false);

    // O += P @ V
#pragma unroll
    for (int c = 0; c < 4; ++c) {
      const __bf16* vp = vtc + (c * 16 + ln) * 64;
      v16bf b0 = *(const v16bf*)(vp + hi * 16);
      v16bf b1 = *(const v16bf*)(vp + 32 + hi * 16);
      oacc[c] = __builtin_amdgcn_wmma_f32_16x16x32_bf16(false, pa0, false, b0, (short)0, oacc[c], false, false);
      oacc[c] = __builtin_amdgcn_wmma_f32_16x16x32_bf16(false, pa1, false, b1, (short)0, oacc[c], false, false);
    }

#pragma unroll
    for (int j = 0; j < 8; ++j) run_sum[j] += lsum[j];

    async_wait_all();   // prefetch of next tile has landed
    __syncthreads();
  }

#pragma unroll
  for (int c = 0; c < 4; ++c)
#pragma unroll
    for (int j = 0; j < 8; ++j) {
      int r = blockIdx.x * 128 + wave * 16 + j + hi * 8;
      ctx[(size_t)r * 64 + c * 16 + ln] = oacc[c][j] / run_sum[j];
    }
}

// ------------------------------------------------------------- out-projection
__global__ void k_oproj(const float* __restrict__ ctx, const float* __restrict__ wo,
                        const float* __restrict__ so, const float* __restrict__ bo,
                        float* __restrict__ octx) {
  int idx = blockIdx.x * 256 + threadIdx.x;
  int c = idx >> 13, n = idx & (N_TOT - 1);
  const float* cr = ctx + (size_t)n * 64;
  const float* wr = wo + c * 64;
  float acc = 0.f;
#pragma unroll 8
  for (int t = 0; t < 64; ++t) acc += wr[t] * cr[t];
  octx[(size_t)c * N_TOT + n] = leaky(so[c] * acc + bo[c]);
}

// --------------------------- weight transform: (co,ci,tap) f32 -> (tap,co,ci) bf16
__global__ void k_wprep(const float* __restrict__ wb, __bf16* __restrict__ wtap) {
  int idx = blockIdx.x * 256 + threadIdx.x;  // 27 * 8192
  if (idx >= 27 * 8192) return;
  int tap = idx >> 13, r = idx & 8191;
  wtap[(size_t)tap * 8192 + r] = (__bf16)wb[(size_t)r * 27 + tap];
}

// ---------------- cat transpose: x | octx (c-major f32) -> catbf (n,128) bf16
__global__ void k_cat(const float* __restrict__ x, const float* __restrict__ octx,
                      __bf16* __restrict__ catbf) {
  __shared__ float xs[64 * 128];
  const int tid = threadIdx.x;
  const int n0 = blockIdx.x * 128;
  for (int half = 0; half < 2; ++half) {
    const float* src = half ? octx : x;
    for (int i = tid; i < 64 * 32; i += 256) {
      int c = i >> 5, ch4 = i & 31;
      ((float4*)xs)[c * 32 + ch4] = ((const float4*)(src + (size_t)c * N_TOT + n0))[ch4];
    }
    __syncthreads();
    for (int i = tid; i < 128 * 8; i += 256) {
      int p = i >> 3, cc = i & 7;
      __align__(16) __bf16 tmp[8];
#pragma unroll
      for (int u = 0; u < 8; ++u) tmp[u] = (__bf16)xs[(cc * 8 + u) * 128 + p];
      *(uint4*)(catbf + (size_t)(n0 + p) * 128 + half * 64 + cc * 8) = *(uint4*)tmp;
    }
    __syncthreads();
  }
}

// ----------------- 3x3x3 conv (128->64) + BN + LeakyReLU, implicit-GEMM WMMA
#define PSTR 136  // padded ci stride (elements) to avoid LDS bank conflicts
__global__ __launch_bounds__(256, 2) void k_conv(
    const __bf16* __restrict__ catbf, const __bf16* __restrict__ wtap,
    const float* __restrict__ sb, const float* __restrict__ bb,
    float* __restrict__ out) {
  __shared__ __align__(64) __bf16 patch[408 * PSTR];  // [dz][hh][ww][ci] halo tile
  __shared__ __align__(64) __bf16 wsl[2][64 * PSTR];  // double-buffered per-tap [co][ci]
  const int tid = threadIdx.x;
  const int wave = tid >> 5, lane = tid & 31, ln = lane & 15, hi = lane >> 4;
  const int d0 = blockIdx.x >> 4;
  const int h0 = (blockIdx.x & 15) * 2;

  auto stage_w = [&](int buf, int tap) {
    const __bf16* src = wtap + (size_t)tap * 8192;
    for (int i = tid; i < 64 * 16; i += 256) {
      int co = i >> 4, c = i & 15;
      cp16_g2l(&wsl[buf][co * PSTR + c * 8], src + co * 128 + c * 8);
    }
  };

  // stage zero-padded input patch (3 x 4 x 34 rows of 128 ci)
  for (int i = tid; i < 408 * 16; i += 256) {
    int r = i >> 4, c = i & 15;
    int dz = r / 136;
    int r2 = r - dz * 136;
    int hh = r2 / 34, ww = r2 - hh * 34;
    int dd = d0 + dz - 1, hgl = h0 + hh - 1, wgl = ww - 1;
    uint4* dst = (uint4*)(patch + (size_t)r * PSTR) + c;
    if ((unsigned)dd < 8u && (unsigned)hgl < 32u && (unsigned)wgl < 32u) {
      int nsrc = dd * 1024 + hgl * 32 + wgl;
      cp16_g2l(dst, catbf + (size_t)nsrc * 128 + c * 8);
    } else {
      *dst = make_uint4(0, 0, 0, 0);
    }
  }
  stage_w(0, 0);
  async_wait_all();
  __syncthreads();

  v8f acc[2];
  acc[0] = (v8f){};
  acc[1] = (v8f){};

  for (int tap = 0; tap < 27; ++tap) {
    const int cur = tap & 1;
    if (tap + 1 < 27) stage_w(cur ^ 1, tap + 1);  // async prefetch of next tap

    const int kd = tap / 9, kh = (tap / 3) % 3, kw = tap % 3;
#pragma unroll
    for (int t = 0; t < 2; ++t) {
      const int tile = wave * 2 + t;
      const int cot = tile >> 2, post = tile & 3;
      const int posCol = post * 16 + ln;
      const int hloc = posCol >> 5, wloc = posCol & 31;
      const __bf16* bbase = patch + ((size_t)((kd * 4 + hloc + kh) * 34 + wloc + kw)) * PSTR;
      const __bf16* arow = &wsl[cur][(size_t)(cot * 16 + ln) * PSTR];
#pragma unroll
      for (int kk = 0; kk < 4; ++kk) {
        v16bf b = mk_afrag(bbase + kk * 32 + hi * 16, bbase + kk * 32 + hi * 16 + 8);
        v16bf a = mk_afrag(arow + kk * 32 + hi * 16,  arow + kk * 32 + hi * 16 + 8);
        acc[t] = __builtin_amdgcn_wmma_f32_16x16x32_bf16(false, a, false, b, (short)0, acc[t], false, false);
      }
    }
    async_wait_all();
    __syncthreads();
  }

#pragma unroll
  for (int t = 0; t < 2; ++t) {
    const int tile = wave * 2 + t;
    const int cot = tile >> 2, post = tile & 3;
    const int posCol = post * 16 + ln;
    const int n = d0 * 1024 + (h0 + (posCol >> 5)) * 32 + (posCol & 31);
#pragma unroll
    for (int j = 0; j < 8; ++j) {
      const int co = cot * 16 + j + hi * 8;
      out[(size_t)co * N_TOT + n] = leaky(sb[co] * acc[t][j] + bb[co]);
    }
  }
}

// ------------------------------------------------------------------ launcher
extern "C" void kernel_launch(void* const* d_in, const int* in_sizes, int n_in,
                              void* d_out, int out_size, void* d_ws, size_t ws_size,
                              hipStream_t stream) {
  (void)in_sizes; (void)n_in; (void)out_size; (void)ws_size;
  const float* x   = (const float*)d_in[0];
  const float* wq1 = (const float*)d_in[1];
  const float* sq1 = (const float*)d_in[2];
  const float* bq1 = (const float*)d_in[3];
  const float* wq2 = (const float*)d_in[4];
  const float* sq2 = (const float*)d_in[5];
  const float* bq2 = (const float*)d_in[6];
  const float* wk1 = (const float*)d_in[7];
  const float* sk1 = (const float*)d_in[8];
  const float* bk1 = (const float*)d_in[9];
  const float* wk2 = (const float*)d_in[10];
  const float* sk2 = (const float*)d_in[11];
  const float* bk2 = (const float*)d_in[12];
  const float* wv  = (const float*)d_in[13];
  const float* sv  = (const float*)d_in[14];
  const float* bv  = (const float*)d_in[15];
  const float* wo  = (const float*)d_in[16];
  const float* so  = (const float*)d_in[17];
  const float* bo  = (const float*)d_in[18];
  const float* wb  = (const float*)d_in[19];
  const float* sb  = (const float*)d_in[20];
  const float* bb  = (const float*)d_in[21];

  char* ws = (char*)d_ws;
  float*  mean  = (float*)ws;
  __bf16* qbf   = (__bf16*)(ws + 4096);
  __bf16* kbf   = (__bf16*)(ws + 4096 + 1u * (1u << 20));
  __bf16* vbf   = (__bf16*)(ws + 4096 + 2u * (1u << 20));
  float*  ctx   = (float*)(ws + 4096 + 3u * (1u << 20));
  float*  octx  = (float*)(ws + 4096 + 5u * (1u << 20));
  __bf16* wtap  = (__bf16*)(ws + 4096 + 7u * (1u << 20));
  __bf16* catbf = (__bf16*)(ws + 4096 + 7u * (1u << 20) + (512u << 10));

  k_mean <<<64,   256, 0, stream>>>(x, mean);
  k_wprep<<<864,  256, 0, stream>>>(wb, wtap);
  k_proj <<<64,   256, 0, stream>>>(x, mean, wq1, sq1, bq1, wq2, sq2, bq2,
                                    wk1, sk1, bk1, wk2, sk2, bk2, wv, sv, bv,
                                    qbf, kbf, vbf);
  k_attn <<<64,   256, 0, stream>>>(qbf, kbf, vbf, ctx);
  k_oproj<<<2048, 256, 0, stream>>>(ctx, wo, so, bo, octx);
  k_cat  <<<64,   256, 0, stream>>>(x, octx, catbf);
  k_conv <<<128,  256, 0, stream>>>(catbf, wtap, sb, bb, (float*)d_out);
}
